// Net_13477607375097
// MI455X (gfx1250) — matrix operand order repaired and checked by reading, not compiled, not run on hardware
//
#include <hip/hip_runtime.h>
#include <hip/hip_bf16.h>
#include <stdint.h>

#define N_NODES   50000
#define N_EDGES   1600000
#define F_IN      128
#define C_DIM     64
#define MLP_DIM   128
#define N_CLASSES 10
#define N_GRAPHS  512
#define CAT_DIM   192   // 3 * C_DIM

typedef __attribute__((ext_vector_type(16))) _Float16 v16h;
typedef __attribute__((ext_vector_type(8)))  float    v8f;
typedef __attribute__((ext_vector_type(8)))  _Float16 v8h;

union VU   { v16h h; uint32_t u[8]; };        // WMMA A/B fragment <-> 8 packed k-pairs
union U4V8 { uint4 u; v8h h; };               // 16B <-> 8 x f16
union F4H  { uint2 u; _Float16 h[4]; };       // 8B  <-> 4 x f16

static __device__ __forceinline__ uint32_t pack2h(float lo, float hi) {
  union { uint32_t u; _Float16 h[2]; } r;
  r.h[0] = (_Float16)lo; r.h[1] = (_Float16)hi;
  return r.u;
}

static __device__ __forceinline__ void atomic_max_f32(float* addr, float val) {
  // monotonic two-sided trick; buffer initialized to -inf (0xFF800000)
  if (val >= 0.0f) atomicMax((int*)addr, __float_as_int(val));
  else             atomicMin((unsigned int*)addr, __float_as_uint(val));
}

// ---------------------------------------------------------------- fill
__global__ void fill_u32_kernel(uint32_t* __restrict__ p, uint32_t v, int n) {
  int i = blockIdx.x * blockDim.x + threadIdx.x;
  if (i < n) p[i] = v;
}

// ---------------------------------------------------------------- fc0: h = x @ W + b  (WMMA)
__global__ __launch_bounds__(256) void fc0_wmma_kernel(
    const float* __restrict__ x, const float* __restrict__ w,
    const float* __restrict__ bias, float* __restrict__ h) {
  __shared__ uint32_t wPk[F_IN / 2][C_DIM];   // packed k-pairs, 16 KB
  __shared__ uint32_t xU[8][16][F_IN / 2];    // 16 rows x 128 f16 (as u32), 32 KB
  const int tid = threadIdx.x;
  for (int i = tid; i < (F_IN / 2) * C_DIM; i += 256) {
    int kk = i >> 6, n = i & 63;
    wPk[kk][n] = pack2h(w[(2 * kk) * C_DIM + n], w[(2 * kk + 1) * C_DIM + n]);
  }
  const int wave = tid >> 5, lane = tid & 31;
  const int m16 = lane & 15, hi = lane >> 4;
  const int NT = N_NODES / 16;                // 3125
  int tile = blockIdx.x * 8 + wave;
  if (tile >= NT) tile = NT - 1;              // duplicate work, identical writes
  const int rb = tile * 16;
  {
    const int row = lane >> 1, half = lane & 1;
    const float4* xp = (const float4*)(x + (size_t)(rb + row) * F_IN + half * 64);
    for (int q = 0; q < 16; ++q) {
      float4 vv = xp[q];
      F4H t;
      t.h[0] = (_Float16)vv.x; t.h[1] = (_Float16)vv.y;
      t.h[2] = (_Float16)vv.z; t.h[3] = (_Float16)vv.w;
      *(uint2*)&xU[wave][row][half * 32 + q * 2] = t.u;
    }
  }
  __syncthreads();
  v8f acc[4];
  for (int nt = 0; nt < 4; ++nt) {
    float bv = bias[nt * 16 + m16];
    for (int v = 0; v < 8; ++v) acc[nt][v] = bv;
  }
#pragma unroll
  for (int ks = 0; ks < 4; ++ks) {
    VU a;
#pragma unroll
    for (int v = 0; v < 8; ++v)
      a.u[v] = xU[wave][m16][ks * 16 + ((v & 4) ? 8 : 0) + hi * 4 + (v & 3)];
#pragma unroll
    for (int nt = 0; nt < 4; ++nt) {
      VU b;
      int n = nt * 16 + m16;
#pragma unroll
      for (int v = 0; v < 8; ++v)
        b.u[v] = wPk[ks * 16 + ((v & 4) ? 8 : 0) + hi * 4 + (v & 3)][n];
      acc[nt] = __builtin_amdgcn_wmma_f32_16x16x32_f16(
          false, a.h, false, b.h, (short)0, acc[nt], false, false);
    }
  }
  for (int nt = 0; nt < 4; ++nt)
    for (int v = 0; v < 8; ++v)
      h[(size_t)(rb + v + hi * 8) * C_DIM + nt * 16 + m16] = acc[nt][v];
}

// ---------------------------------------------------------------- BN stats (column sum / sumsq)
#define BN_BLOCKS 125
#define BN_ROWS   400
__global__ __launch_bounds__(256) void bn_stats_kernel(
    const float* __restrict__ h, float* __restrict__ sum, float* __restrict__ sumsq) {
  const int c = threadIdx.x & 63, r0 = threadIdx.x >> 6;
  const int start = blockIdx.x * BN_ROWS;
  int end = start + BN_ROWS; if (end > N_NODES) end = N_NODES;
  float s = 0.f, s2 = 0.f;
  for (int r = start + r0; r < end; r += 4) {
    float v = h[(size_t)r * C_DIM + c];
    s += v; s2 += v * v;
  }
  __shared__ float ls[4][64], ls2[4][64];
  ls[r0][c] = s; ls2[r0][c] = s2;
  __syncthreads();
  if (r0 == 0) {
    float t  = ls[0][c]  + ls[1][c]  + ls[2][c]  + ls[3][c];
    float t2 = ls2[0][c] + ls2[1][c] + ls2[2][c] + ls2[3][c];
    atomicAdd(&sum[c], t);
    atomicAdd(&sumsq[c], t2);
  }
}

// ---------------------------------------------------------------- BN apply -> f16 node features
__global__ void bn_apply_kernel(const float* __restrict__ h,
                                const float* __restrict__ sum, const float* __restrict__ sumsq,
                                const float* __restrict__ g, const float* __restrict__ b,
                                _Float16* __restrict__ hbn) {
  int i = blockIdx.x * blockDim.x + threadIdx.x;
  if (i >= N_NODES * C_DIM) return;
  int c = i & 63;
  const float inv_n = 1.0f / (float)N_NODES;
  float mu = sum[c] * inv_n;
  float var = sumsq[c] * inv_n - mu * mu;
  float scale = g[c] * rsqrtf(var + 1e-5f);
  float shift = b[c] - mu * scale;
  hbn[i] = (_Float16)(h[i] * scale + shift);
}

// ---------------------------------------------------------------- EdgeConv (gather + 2-layer MLP via WMMA + max-scatter)
#define EDGE_BLOCKS 1250
__global__ __launch_bounds__(256) void edge_conv_kernel(
    const _Float16* __restrict__ hbn,
    const int* __restrict__ src, const int* __restrict__ dst,
    const float* __restrict__ w1, const float* __restrict__ b1,
    const float* __restrict__ w2, const float* __restrict__ b2,
    float* __restrict__ agg) {
  __shared__ uint32_t w1Pk[64][64];          // packed k-pairs of w1[128][64], 16 KB
  __shared__ uint32_t w2Pk[32][64];          // packed k-pairs of w2[64][64],   8 KB
  __shared__ uint32_t xiU[8][16][32];        // xi tile (f16 pairs), 16 KB; reused as hmid
  __shared__ uint32_t djU[8][16][32];        // xj - xi tile, 16 KB
  __shared__ int dstS[8][16];

  const int tid = threadIdx.x;
  for (int i = tid; i < 64 * 64; i += 256) {
    int kk = i >> 6, n = i & 63;
    w1Pk[kk][n] = pack2h(w1[(2 * kk) * 64 + n], w1[(2 * kk + 1) * 64 + n]);
  }
  for (int i = tid; i < 32 * 64; i += 256) {
    int kk = i >> 6, n = i & 63;
    w2Pk[kk][n] = pack2h(w2[(2 * kk) * 64 + n], w2[(2 * kk + 1) * 64 + n]);
  }

  const int wave = tid >> 5, lane = tid & 31;
  const int m16 = lane & 15, hi = lane >> 4;
  const int NTILES = N_EDGES / 16;           // 100000
  const int nslots = EDGE_BLOCKS * 8;        // 10000 wave slots
  const int slot = blockIdx.x * 8 + wave;
  const int iters = (NTILES + nslots - 1) / nslots;   // uniform trip count

  const int grow = lane >> 1, ghalf = lane & 1;       // gather role: 2 lanes per edge

  for (int it = 0; it < iters; ++it) {
    int tile = it * nslots + slot;
    if (tile >= NTILES) tile = NTILES - 1;  // idempotent under max-aggregation
    const int ebase = tile * 16;

    // ---- gather: xi = hbn[dst], dj = hbn[src] - xi (f16, packed) ----
    {
      int e = ebase + grow;
      int s = src[e], d = dst[e];
      if (ghalf == 0) dstS[wave][grow] = d;
      const uint4* xip = (const uint4*)(hbn + (size_t)d * C_DIM + ghalf * 32);
      const uint4* xjp = (const uint4*)(hbn + (size_t)s * C_DIM + ghalf * 32);
#pragma unroll
      for (int q = 0; q < 4; ++q) {
        U4V8 ui, uj, ud;
        ui.u = xip[q]; uj.u = xjp[q];
        ud.h = uj.h - ui.h;                  // v_pk_sub_f16
        *(uint4*)&xiU[wave][grow][ghalf * 16 + q * 4] = ui.u;
        *(uint4*)&djU[wave][grow][ghalf * 16 + q * 4] = ud.u;
      }
      if (it + 1 < iters) {                  // prefetch next tile's edge indices
        int tn = (it + 1) * nslots + slot;
        if (tn >= NTILES) tn = NTILES - 1;
        __builtin_prefetch(&src[tn * 16], 0, 0);
        __builtin_prefetch(&dst[tn * 16], 0, 0);
      }
    }
    __syncthreads();

    // ---- layer 1: feat[16x128] @ w1[128x64] + b1 ----
    v8f acc[4];
    for (int nt = 0; nt < 4; ++nt) {
      float bv = b1[nt * 16 + m16];
      for (int v = 0; v < 8; ++v) acc[nt][v] = bv;
    }
#pragma unroll
    for (int ks = 0; ks < 4; ++ks) {
      const uint32_t(*featU)[32] = (ks < 2) ? xiU[wave] : djU[wave];
      VU a;
#pragma unroll
      for (int v = 0; v < 8; ++v)
        a.u[v] = featU[m16][(ks & 1) * 16 + ((v & 4) ? 8 : 0) + hi * 4 + (v & 3)];
#pragma unroll
      for (int nt = 0; nt < 4; ++nt) {
        VU b;
        int n = nt * 16 + m16;
#pragma unroll
        for (int v = 0; v < 8; ++v)
          b.u[v] = w1Pk[ks * 16 + ((v & 4) ? 8 : 0) + hi * 4 + (v & 3)][n];
        acc[nt] = __builtin_amdgcn_wmma_f32_16x16x32_f16(
            false, a.h, false, b.h, (short)0, acc[nt], false, false);
      }
    }
    // relu -> hmid in LDS (reuse xiU slice); D layout: M = v + hi*8, N = lane&15
    {
      _Float16* hm = (_Float16*)xiU[wave];
      for (int nt = 0; nt < 4; ++nt)
        for (int v = 0; v < 8; ++v)
          hm[(v + hi * 8) * 64 + nt * 16 + m16] = (_Float16)fmaxf(acc[nt][v], 0.0f);
    }
    __syncthreads();

    // ---- layer 2: hmid[16x64] @ w2[64x64] + b2 ----
    v8f acc2[4];
    for (int nt = 0; nt < 4; ++nt) {
      float bv = b2[nt * 16 + m16];
      for (int v = 0; v < 8; ++v) acc2[nt][v] = bv;
    }
#pragma unroll
    for (int ks = 0; ks < 2; ++ks) {
      VU a;
#pragma unroll
      for (int v = 0; v < 8; ++v)
        a.u[v] = xiU[wave][m16][ks * 16 + ((v & 4) ? 8 : 0) + hi * 4 + (v & 3)];
#pragma unroll
      for (int nt = 0; nt < 4; ++nt) {
        VU b;
        int n = nt * 16 + m16;
#pragma unroll
        for (int v = 0; v < 8; ++v)
          b.u[v] = w2Pk[ks * 16 + ((v & 4) ? 8 : 0) + hi * 4 + (v & 3)][n];
        acc2[nt] = __builtin_amdgcn_wmma_f32_16x16x32_f16(
            false, a.h, false, b.h, (short)0, acc2[nt], false, false);
      }
    }

    // ---- segment-max scatter ----
    for (int nt = 0; nt < 4; ++nt) {
      int col = nt * 16 + m16;
      for (int v = 0; v < 8; ++v) {
        int node = dstS[wave][v + hi * 8];
        atomic_max_f32(&agg[(size_t)node * C_DIM + col], acc2[nt][v]);
      }
    }
    __syncthreads();   // protect LDS tiles (WAR) before next iteration's gather
  }
}

// ---------------------------------------------------------------- post: h = max(agg,0) (handles -inf), write cat slice
__global__ void post_kernel(const float* __restrict__ agg, float* __restrict__ h,
                            float* __restrict__ cat, int blk) {
  int i = blockIdx.x * blockDim.x + threadIdx.x;
  if (i >= N_NODES * C_DIM) return;
  int n = i >> 6, c = i & 63;
  float v = fmaxf(agg[i], 0.0f);   // max(-inf,0)=0 == where(isfinite)->relu
  h[i] = v;
  cat[(size_t)n * CAT_DIM + blk * C_DIM + c] = v;
}

// ---------------------------------------------------------------- mean pool accumulation
__global__ void pool_kernel(const float* __restrict__ cat, const int* __restrict__ batch,
                            float* __restrict__ sums, float* __restrict__ cnt) {
  int i = blockIdx.x * blockDim.x + threadIdx.x;
  if (i >= N_NODES * CAT_DIM) return;
  int n = i / CAT_DIM, c = i - n * CAT_DIM;
  int g = batch[n];
  atomicAdd(&sums[(size_t)g * CAT_DIM + c], cat[i]);
  if (c == 0) atomicAdd(&cnt[g], 1.0f);
}

// ---------------------------------------------------------------- head: mean, fc1+relu, fc2, log_softmax
__global__ __launch_bounds__(192) void head_kernel(
    const float* __restrict__ sums, const float* __restrict__ cnt,
    const float* __restrict__ fc1w, const float* __restrict__ fc1b,
    const float* __restrict__ fc2w, const float* __restrict__ fc2b,
    float* __restrict__ out) {
  const int g = blockIdx.x, t = threadIdx.x;
  __shared__ float pooled[CAT_DIM];
  __shared__ float hid[MLP_DIM];
  __shared__ float logits[N_CLASSES];
  float cdiv = 1.0f / fmaxf(cnt[g], 1.0f);
  if (t < CAT_DIM) pooled[t] = sums[(size_t)g * CAT_DIM + t] * cdiv;
  __syncthreads();
  if (t < MLP_DIM) {
    float a = fc1b[t];
    for (int k = 0; k < CAT_DIM; ++k) a += pooled[k] * fc1w[k * MLP_DIM + t];
    hid[t] = fmaxf(a, 0.0f);
  }
  __syncthreads();
  if (t < N_CLASSES) {
    float a = fc2b[t];
    for (int k = 0; k < MLP_DIM; ++k) a += hid[k] * fc2w[k * N_CLASSES + t];
    logits[t] = a;
  }
  __syncthreads();
  if (t == 0) {
    float mx = logits[0];
    for (int j = 1; j < N_CLASSES; ++j) mx = fmaxf(mx, logits[j]);
    float se = 0.0f;
    for (int j = 0; j < N_CLASSES; ++j) se += expf(logits[j] - mx);
    float lse = mx + logf(se);
    for (int j = 0; j < N_CLASSES; ++j) out[(size_t)g * N_CLASSES + j] = logits[j] - lse;
  }
}

// ================================================================ launch
extern "C" void kernel_launch(void* const* d_in, const int* in_sizes, int n_in,
                              void* d_out, int out_size, void* d_ws, size_t ws_size,
                              hipStream_t stream) {
  const float* x       = (const float*)d_in[0];
  const int*   eidx    = (const int*)d_in[1];
  const int*   batch   = (const int*)d_in[2];
  const float* fc0_w   = (const float*)d_in[3];
  const float* fc0_b   = (const float*)d_in[4];
  const float* fc1_w   = (const float*)d_in[5];
  const float* fc1_b   = (const float*)d_in[6];
  const float* fc2_w   = (const float*)d_in[7];
  const float* fc2_b   = (const float*)d_in[8];
  const float* bn_g[3] = {(const float*)d_in[9],  (const float*)d_in[15], (const float*)d_in[21]};
  const float* bn_b[3] = {(const float*)d_in[10], (const float*)d_in[16], (const float*)d_in[22]};
  const float* w1[3]   = {(const float*)d_in[11], (const float*)d_in[17], (const float*)d_in[23]};
  const float* b1[3]   = {(const float*)d_in[12], (const float*)d_in[18], (const float*)d_in[24]};
  const float* w2[3]   = {(const float*)d_in[13], (const float*)d_in[19], (const float*)d_in[25]};
  const float* b2[3]   = {(const float*)d_in[14], (const float*)d_in[20], (const float*)d_in[26]};
  const int* src = eidx;
  const int* dst = eidx + N_EDGES;

  // workspace carve (256B aligned)
  char* ws = (char*)d_ws;
  size_t off = 0;
  auto carve = [&](size_t bytes) -> char* {
    char* p = ws + off;
    off = (off + bytes + 255) & ~(size_t)255;
    return p;
  };
  float*     h       = (float*)    carve((size_t)N_NODES * C_DIM * 4);
  _Float16*  hbn     = (_Float16*) carve((size_t)N_NODES * C_DIM * 2);
  float*     agg     = (float*)    carve((size_t)N_NODES * C_DIM * 4);
  float*     cat     = (float*)    carve((size_t)N_NODES * CAT_DIM * 4);
  float*     bn_sum  = (float*)    carve(128 * 4);          // sum[64] ++ sumsq[64]
  float*     bn_sq   = bn_sum + 64;
  float*     pool    = (float*)    carve((size_t)N_GRAPHS * (CAT_DIM + 1) * 4); // sums ++ cnt
  float*     pcnt    = pool + (size_t)N_GRAPHS * CAT_DIM;

  const int NC64  = N_NODES * C_DIM;        // 3,200,000
  const int NC192 = N_NODES * CAT_DIM;      // 9,600,000

  fc0_wmma_kernel<<<(N_NODES / 16 + 7) / 8, 256, 0, stream>>>(x, fc0_w, fc0_b, h);

  for (int i = 0; i < 3; ++i) {
    fill_u32_kernel<<<1, 256, 0, stream>>>((uint32_t*)bn_sum, 0u, 128);
    bn_stats_kernel<<<BN_BLOCKS, 256, 0, stream>>>(h, bn_sum, bn_sq);
    bn_apply_kernel<<<(NC64 + 255) / 256, 256, 0, stream>>>(h, bn_sum, bn_sq,
                                                            bn_g[i], bn_b[i], hbn);
    fill_u32_kernel<<<(NC64 + 255) / 256, 256, 0, stream>>>((uint32_t*)agg, 0xFF800000u, NC64);
    edge_conv_kernel<<<EDGE_BLOCKS, 256, 0, stream>>>(
        hbn, src, dst, w1[i], b1[i], w2[i], b2[i], agg);
    post_kernel<<<(NC64 + 255) / 256, 256, 0, stream>>>(agg, h, cat, i);
  }

  fill_u32_kernel<<<(N_GRAPHS * (CAT_DIM + 1) + 255) / 256, 256, 0, stream>>>(
      (uint32_t*)pool, 0u, N_GRAPHS * (CAT_DIM + 1));
  pool_kernel<<<(NC192 + 255) / 256, 256, 0, stream>>>(cat, batch, pool, pcnt);
  head_kernel<<<N_GRAPHS, 192, 0, stream>>>(pool, pcnt, fc1_w, fc1_b, fc2_w, fc2_b,
                                            (float*)d_out);
}
